// TempProductKeyRouter_9620726743478
// MI455X (gfx1250) — matrix operand order repaired and checked by reading, not compiled, not run on hardware
//
#include <hip/hip_runtime.h>
#include <hip/hip_bf16.h>
#include <math.h>

typedef __attribute__((ext_vector_type(16))) __bf16 v16bf;
typedef __attribute__((ext_vector_type(8)))  float  v8f;
typedef __attribute__((ext_vector_type(4)))  float  v4f;

#define D_DIM 4096
#define SK    128
#define TOPK  16
#define NTOK  16384   // 2 * 8192
#define MT    16      // tokens per workgroup (one WMMA M tile)

// descending-order top-16 insertion; fully unrolled so arrays stay in VGPRs
#define INSERT16(VV, II, val, idx)                                         \
  do {                                                                     \
    if ((val) > VV[15]) {                                                  \
      VV[15] = (val); II[15] = (idx);                                      \
      _Pragma("unroll")                                                    \
      for (int _q = 15; _q >= 1; --_q) {                                   \
        if (VV[_q] > VV[_q - 1]) {                                         \
          float _tv = VV[_q]; VV[_q] = VV[_q - 1]; VV[_q - 1] = _tv;       \
          int   _ti = II[_q]; II[_q] = II[_q - 1]; II[_q - 1] = _ti;       \
        }                                                                  \
      }                                                                    \
    }                                                                      \
  } while (0)

__global__ __launch_bounds__(256) void pk_convert_w(const float* __restrict__ W1,
                                                    const float* __restrict__ W2,
                                                    __bf16* __restrict__ W1b,
                                                    __bf16* __restrict__ W2b, int n) {
  for (int i = blockIdx.x * blockDim.x + threadIdx.x; i < n;
       i += gridDim.x * blockDim.x) {
    W1b[i] = (__bf16)W1[i];
    W2b[i] = (__bf16)W2[i];
  }
}

__global__ __launch_bounds__(256) void pk_router(const float* __restrict__ x,
                                                 const __bf16* __restrict__ W1b,
                                                 const __bf16* __restrict__ W2b,
                                                 const float* __restrict__ log_tau,
                                                 float* __restrict__ out) {
  __shared__ float sS[MT * 256];        // [token][col] cols 0..127 = s1, 128..255 = s2
  __shared__ float sTa[MT][16], sTb[MT][16];
  __shared__ int   sIa[MT][16], sIb[MT][16];

  const int tid     = threadIdx.x;
  const int lane    = tid & 31;
  const int wid     = tid >> 5;
  const int tokBase = blockIdx.x * MT;

  // ---- Phase 1: s1/s2 GEMM via bf16 WMMA (D = A x B + C, fp32 accumulate) ----
  const int nt0 = wid * 2;              // N tiles 0..15 over combined 256 columns
  const int nt1 = wid * 2 + 1;
  const __bf16* w0 = (nt0 < 8) ? W1b : W2b;
  const __bf16* w1 = (nt1 < 8) ? W1b : W2b;
  const int c0 = (nt0 & 7) * 16 + (lane & 15);   // weight row (output column)
  const int c1 = (nt1 & 7) * 16 + (lane & 15);
  const int kgroup = (lane < 16) ? 0 : 16;       // B layout: lanes 0-15 K 0..15, 16-31 K 16..31
  const int kbase  = (lane < 16) ? 0 : 16;       // A layout: lanes 0-15 K 0..15, 16-31 K 16..31
  const int mrow   = lane & 15;

  const float*  xr  = x  + (size_t)(tokBase + mrow) * D_DIM;
  const __bf16* wr0 = w0 + (size_t)c0 * D_DIM + kgroup;
  const __bf16* wr1 = w1 + (size_t)c1 * D_DIM + kgroup;

  v8f acc0 = {0.f, 0.f, 0.f, 0.f, 0.f, 0.f, 0.f, 0.f};
  v8f acc1 = {0.f, 0.f, 0.f, 0.f, 0.f, 0.f, 0.f, 0.f};

  for (int kk = 0; kk < D_DIM; kk += 32) {
    if (wid == 0 && kk + 256 < D_DIM)
      __builtin_prefetch(xr + kk + 256, 0, 3);   // global_prefetch_b8, x tile ahead

    v4f f0 = *(const v4f*)(xr + kk + kbase);
    v4f f1 = *(const v4f*)(xr + kk + kbase + 4);
    v4f f2 = *(const v4f*)(xr + kk + kbase + 8);
    v4f f3 = *(const v4f*)(xr + kk + kbase + 12);
    v16bf a;
    a[0]  = (__bf16)f0[0]; a[1]  = (__bf16)f0[1]; a[2]  = (__bf16)f0[2]; a[3]  = (__bf16)f0[3];
    a[4]  = (__bf16)f1[0]; a[5]  = (__bf16)f1[1]; a[6]  = (__bf16)f1[2]; a[7]  = (__bf16)f1[3];
    a[8]  = (__bf16)f2[0]; a[9]  = (__bf16)f2[1]; a[10] = (__bf16)f2[2]; a[11] = (__bf16)f2[3];
    a[12] = (__bf16)f3[0]; a[13] = (__bf16)f3[1]; a[14] = (__bf16)f3[2]; a[15] = (__bf16)f3[3];

    v16bf b0 = *(const v16bf*)(wr0 + kk);
    v16bf b1 = *(const v16bf*)(wr1 + kk);

    acc0 = __builtin_amdgcn_wmma_f32_16x16x32_bf16(false, a, false, b0,
                                                   (short)0, acc0, false, false);
    acc1 = __builtin_amdgcn_wmma_f32_16x16x32_bf16(false, a, false, b1,
                                                   (short)0, acc1, false, false);
  }

  // C/D layout: VGPR r, lanes 0-15 -> M=r, lanes 16-31 -> M=r+8; N = lane&15
  {
    const int half = lane >> 4;
    const int nloc = lane & 15;
#pragma unroll
    for (int r = 0; r < 8; ++r) {
      const int m = r + 8 * half;
      sS[m * 256 + nt0 * 16 + nloc] = acc0[r];
      sS[m * 256 + nt1 * 16 + nloc] = acc1[r];
    }
  }
  __syncthreads();

  // ---- Phase 2: stream full scores (outer sum), non-temporal float4 stores ----
  {
    float* scoresOut = out + (size_t)2 * NTOK * TOPK + (size_t)tokBase * (SK * SK);
    const int total = MT * SK * SK;      // 262144 floats per workgroup
    for (int idx = tid * 4; idx < total; idx += 256 * 4) {
      const int t   = idx >> 14;         // / 16384
      const int rem = idx & 16383;
      const int i   = rem >> 7;          // s1 index
      const int j   = rem & 127;         // s2 index (multiple of 4)
      const float s1v = sS[t * 256 + i];
      v4f s2v = *(const v4f*)&sS[t * 256 + 128 + j];
      v4f o   = s2v + s1v;
      __builtin_nontemporal_store(o, (v4f*)(scoresOut + (size_t)t * (SK * SK) + rem));
    }
  }
  __syncthreads();

  // ---- Phase 3: product-key top-16 + softmax (one thread per token) ----
  if (tid < MT) {
    const int t = tid;
    float av[16], bv[16];
    int   ai[16], bi[16];
#pragma unroll
    for (int q = 0; q < 16; ++q) {
      av[q] = -3.0e38f; bv[q] = -3.0e38f; ai[q] = 0; bi[q] = 0;
    }
    for (int i = 0; i < SK; ++i) {
      float v = sS[t * 256 + i];
      INSERT16(av, ai, v, i);
      float u = sS[t * 256 + 128 + i];
      INSERT16(bv, bi, u, i);
    }
#pragma unroll
    for (int q = 0; q < 16; ++q) {
      sTa[t][q] = av[q]; sIa[t][q] = ai[q];
      sTb[t][q] = bv[q]; sIb[t][q] = bi[q];
    }
    // merge 16x16 candidates; both lists descending -> early break
    float cv[16]; int ci[16];
#pragma unroll
    for (int q = 0; q < 16; ++q) { cv[q] = -3.0e38f; ci[q] = 0; }
    for (int i = 0; i < 16; ++i) {
      const float a_  = sTa[t][i];
      const int   ia_ = sIa[t][i];
      if (a_ + sTb[t][0] <= cv[15]) break;
      for (int j = 0; j < 16; ++j) {
        const float v = a_ + sTb[t][j];
        if (v <= cv[15]) break;
        const int gidx = ia_ * SK + sIb[t][j];
        INSERT16(cv, ci, v, gidx);
      }
    }
    // softmax(topk_vals / tau)
    const float tau     = expf(log_tau[0]);
    const float inv_tau = 1.0f / tau;
    const float mx = cv[0];
    float e[16], sum = 0.0f;
#pragma unroll
    for (int q = 0; q < 16; ++q) { e[q] = expf((cv[q] - mx) * inv_tau); sum += e[q]; }
    const float inv_sum = 1.0f / sum;
    const size_t tg = (size_t)(tokBase + t);
#pragma unroll
    for (int q = 0; q < 16; ++q) {
      out[tg * TOPK + q]                        = (float)ci[q];     // topk_idx
      out[(size_t)NTOK * TOPK + tg * TOPK + q]  = e[q] * inv_sum;   // gates
    }
  }
}

extern "C" void kernel_launch(void* const* d_in, const int* in_sizes, int n_in,
                              void* d_out, int out_size, void* d_ws, size_t ws_size,
                              hipStream_t stream) {
  (void)in_sizes; (void)n_in; (void)out_size; (void)ws_size;
  const float* x       = (const float*)d_in[0];
  const float* W1      = (const float*)d_in[1];
  const float* W2      = (const float*)d_in[2];
  const float* log_tau = (const float*)d_in[3];

  __bf16* W1b = (__bf16*)d_ws;                 // 1 MB
  __bf16* W2b = W1b + (size_t)SK * D_DIM;      // 1 MB  (ws usage: 2 MB)
  const int n = SK * D_DIM;

  pk_convert_w<<<(n + 255) / 256, 256, 0, stream>>>(W1, W2, W1b, W2b, n);
  pk_router<<<NTOK / MT, 256, 0, stream>>>(x, W1b, W2b, log_tau, (float*)d_out);
}
